// LGCore_23613730193937
// MI455X (gfx1250) — matrix operand order repaired and checked by reading, not compiled, not run on hardware
//
#include <hip/hip_runtime.h>
#include <hip/hip_bf16.h>

#define N_NODES 16384
#define M_NODES 8192
#define NEDGE   524288
#define DFEAT   128
#define LN_EPS  1e-5f

typedef float  v8f  __attribute__((ext_vector_type(8)));
typedef float  v2f  __attribute__((ext_vector_type(2)));
typedef __bf16 v16bf __attribute__((ext_vector_type(16)));

// Pack two fp32 into two bf16 (round via +0x8000) with a single v_perm_b32.
__device__ __forceinline__ unsigned pack_bf16(float lo, float hi) {
    unsigned a = __float_as_uint(lo) + 0x8000u;
    unsigned b = __float_as_uint(hi) + 0x8000u;
    return __builtin_amdgcn_perm(b, a, 0x07060302u);
}

// gfx1250 async global->LDS copy, 16B per lane, tracked by ASYNCcnt.
__device__ __forceinline__ void async_b128(unsigned lds_off, unsigned voff,
                                           const void* sbase) {
    unsigned long long s = (unsigned long long)(uintptr_t)sbase;
    asm volatile("global_load_async_to_lds_b128 %0, %1, %2"
                 :: "v"(lds_off), "v"(voff), "s"(s) : "memory");
}
__device__ __forceinline__ void wait_async0() {
    asm volatile("s_wait_asynccnt 0x0" ::: "memory");
}

// ---------------- degree / norm kernels ----------------
__global__ void k_zero(float* __restrict__ p, int n) {
    int i = blockIdx.x * blockDim.x + threadIdx.x;
    if (i < n) p[i] = 0.f;
}

__global__ void k_deg(const int* __restrict__ src, const int* __restrict__ dst,
                      float* __restrict__ degO, float* __restrict__ degI) {
    int e = blockIdx.x * blockDim.x + threadIdx.x;
    if (e < NEDGE) {
        atomicAdd(&degO[src[e]], 1.f);
        atomicAdd(&degI[dst[e]], 1.f);
    }
}

__global__ void k_rs(const float* __restrict__ degO, const float* __restrict__ degI,
                     float* __restrict__ rsO, float* __restrict__ rsI) {
    int i = blockIdx.x * blockDim.x + threadIdx.x;
    if (i < N_NODES) {
        rsO[i] = rsqrtf(degO[i] + 1.f);
        rsI[i] = rsqrtf(degI[i] + 1.f);
    }
}

// ---------------- next_h -> bf16, transposed [D][M] ----------------
__global__ void k_bT(const float* __restrict__ nh, unsigned short* __restrict__ bT) {
    int idx = blockIdx.x * blockDim.x + threadIdx.x;   // M*D elements
    int r = idx >> 7;          // row in next_h (K index)
    int c = idx & 127;         // feature (output column)
    unsigned u = __float_as_uint(nh[idx]) + 0x8000u;
    bT[(size_t)c * M_NODES + r] = (unsigned short)(u >> 16);
}

// ---------------- dominant GEMM: fused = curr_inc @ next_h (bf16 WMMA) ----------
// Block = 8 waves = 128 rows x 128 cols. B chunk (8KB: 128 cols x 32 K) is shared
// by all waves -> staged into LDS with async copies, double buffered. A stream is
// software-pipelined in registers with an unroll-by-2 ping-pong.
__global__ __launch_bounds__(256) void k_dense(const float* __restrict__ inc,
                                               const unsigned short* __restrict__ bT,
                                               float* __restrict__ fused) {
    __shared__ unsigned char lbuf[2][8192];
    const int tid  = threadIdx.x;
    const int wave = tid >> 5;
    const int lane = tid & 31;
    const int m = lane & 15;          // A row within tile / B column within tile
    const int h = lane >> 4;          // lane half
    const int row0 = blockIdx.x * 128 + wave * 16;
    const float* arow = inc + (size_t)(row0 + m) * M_NODES;
    const unsigned lds_base = (unsigned)(uintptr_t)&lbuf[0][0];

    // issue async copy of B chunk [cols 0..127][K k0..k0+31] into buffer b
    auto issueB = [&](int k0, int b) {
        unsigned lb = lds_base + (unsigned)b * 8192u;
        int j = tid;
#pragma unroll
        for (int rep = 0; rep < 2; ++rep) {
            int c = j >> 2, sub = j & 3;
            unsigned voff = (unsigned)(c * (M_NODES * 2) + k0 * 2 + sub * 16);
            async_b128(lb + (unsigned)(c * 64 + sub * 16), voff, bT);
            j += 256;
        }
    };
    auto loadA = [&](float4* ar, int k0) {
        ar[0] = *(const float4*)(arow + k0 + h * 8);
        ar[1] = *(const float4*)(arow + k0 + h * 8 + 4);
        ar[2] = *(const float4*)(arow + k0 + 16 + h * 8);
        ar[3] = *(const float4*)(arow + k0 + 16 + h * 8 + 4);
        __builtin_prefetch(arow + k0 + 256, 0, 0);
    };

    v8f c[8];
#pragma unroll
    for (int t = 0; t < 8; ++t)
#pragma unroll
        for (int i = 0; i < 8; ++i) c[t][i] = 0.f;

    auto packA = [&](const float4* ar) {
        union { unsigned u[8]; v16bf v; } A;
        A.u[0] = pack_bf16(ar[0].x, ar[0].y); A.u[1] = pack_bf16(ar[0].z, ar[0].w);
        A.u[2] = pack_bf16(ar[1].x, ar[1].y); A.u[3] = pack_bf16(ar[1].z, ar[1].w);
        A.u[4] = pack_bf16(ar[2].x, ar[2].y); A.u[5] = pack_bf16(ar[2].z, ar[2].w);
        A.u[6] = pack_bf16(ar[3].x, ar[3].y); A.u[7] = pack_bf16(ar[3].z, ar[3].w);
        return A.v;
    };
    auto compute = [&](v16bf Afr, const unsigned char* buf) {
#pragma unroll
        for (int t = 0; t < 8; ++t) {
            const uint4* bp = (const uint4*)(buf + (t * 16 + m) * 64 + h * 16);
            union { uint4 q[2]; v16bf v; } B;
            B.q[0] = bp[0];
            B.q[1] = bp[2];
            c[t] = __builtin_amdgcn_wmma_f32_16x16x32_bf16(
                false, Afr, false, B.v, (short)0, c[t], false, false);
        }
    };

    float4 arA[4], arB[4];
    loadA(arA, 0);
    issueB(0, 0);

    for (int k0 = 0; k0 < M_NODES; k0 += 64) {
        // ---- even chunk: buffer 0 ----
        wait_async0();
        __syncthreads();
        issueB(k0 + 32, 1);                 // k0+32 <= 8160 always valid
        v16bf Af = packA(arA);
        loadA(arB, k0 + 32);
        compute(Af, lbuf[0]);
        // ---- odd chunk: buffer 1 ----
        wait_async0();
        __syncthreads();
        if (k0 + 64 < M_NODES) {
            issueB(k0 + 64, 0);
        }
        Af = packA(arB);
        if (k0 + 64 < M_NODES) {
            loadA(arA, k0 + 64);
        }
        compute(Af, lbuf[1]);
    }

    // store: VGPR i -> row (h?8:0)+i, col = t*16 + m
    int rbase = row0 + h * 8;
#pragma unroll
    for (int t = 0; t < 8; ++t)
#pragma unroll
        for (int i = 0; i < 8; ++i)
            fused[(size_t)(rbase + i) * DFEAT + t * 16 + m] = c[t][i];
}

// ---------------- SpMM init: acc = x * rsqrt(out_deg) (self-loop term) --------
__global__ void k_initacc(const float* __restrict__ curr_h, const float* __restrict__ fused,
                          const float* __restrict__ rsO,
                          float* __restrict__ acc1, float* __restrict__ acc2) {
    size_t idx = (size_t)blockIdx.x * blockDim.x + threadIdx.x;  // N*D
    int i = (int)(idx >> 7);
    float r = rsO[i];
    acc1[idx] = curr_h[idx] * r;
    acc2[idx] = fused[idx] * r;
}

// ---------------- SpMM edge scatter ----------------
__global__ void k_scatter(const int* __restrict__ src, const int* __restrict__ dst,
                          const float* __restrict__ curr_h, const float* __restrict__ fused,
                          const float* __restrict__ rsO,
                          float* __restrict__ acc1, float* __restrict__ acc2) {
    size_t idx = (size_t)blockIdx.x * blockDim.x + threadIdx.x;  // E*D
    int e = (int)(idx >> 7);
    int d = (int)(idx & 127);
    int s = src[e], t = dst[e];
    float r = rsO[s];
    atomicAdd(&acc1[(size_t)t * DFEAT + d], curr_h[(size_t)s * DFEAT + d] * r);
    atomicAdd(&acc2[(size_t)t * DFEAT + d], fused[(size_t)s * DFEAT + d] * r);
}

// ---------------- projection GEMMs + combine (f32 WMMA 16x16x4) ----------------
// res = 0.5*((agg1*rsI)@Wc + bc)*conv_w + 0.5*((agg2*rsI)@Wf + bf)*td_w
__global__ __launch_bounds__(256) void k_out(const float* __restrict__ acc1,
                                             const float* __restrict__ acc2,
                                             const float* __restrict__ rsI,
                                             const float* __restrict__ Wc,
                                             const float* __restrict__ bc,
                                             const float* __restrict__ Wf,
                                             const float* __restrict__ bf,
                                             const float* __restrict__ cw,
                                             const float* __restrict__ tw,
                                             float* __restrict__ out) {
    int wave = (blockIdx.x * blockDim.x + threadIdx.x) >> 5;
    int lane = threadIdx.x & 31;
    int m = lane & 15, h = lane >> 4;
    int row0 = wave * 16;
    float rin = rsI[row0 + m];
    const float* a1row = acc1 + (size_t)(row0 + m) * DFEAT;
    const float* a2row = acc2 + (size_t)(row0 + m) * DFEAT;

    v8f cc[8], cf[8];
#pragma unroll
    for (int t = 0; t < 8; ++t)
#pragma unroll
        for (int i = 0; i < 8; ++i) { cc[t][i] = 0.f; cf[t][i] = 0.f; }

#pragma unroll 4
    for (int k0 = 0; k0 < DFEAT; k0 += 4) {
        // A 16x4 f32: lanes0-15 K=k0,k0+1; lanes16-31 K=k0+2,k0+3
        float2 t1 = *(const float2*)(a1row + k0 + h * 2);
        float2 t2 = *(const float2*)(a2row + k0 + h * 2);
        v2f A1; A1[0] = t1.x * rin; A1[1] = t1.y * rin;
        v2f A2; A2[0] = t2.x * rin; A2[1] = t2.y * rin;
#pragma unroll
        for (int t = 0; t < 8; ++t) {
            int n = t * 16 + m;
            v2f B1, B2;
            B1[0] = Wc[(k0 + h * 2) * DFEAT + n];
            B1[1] = Wc[(k0 + h * 2 + 1) * DFEAT + n];
            B2[0] = Wf[(k0 + h * 2) * DFEAT + n];
            B2[1] = Wf[(k0 + h * 2 + 1) * DFEAT + n];
            cc[t] = __builtin_amdgcn_wmma_f32_16x16x4_f32(
                false, A1, false, B1, (short)0, cc[t], false, false);
            cf[t] = __builtin_amdgcn_wmma_f32_16x16x4_f32(
                false, A2, false, B2, (short)0, cf[t], false, false);
        }
    }
    int rbase = row0 + h * 8;
#pragma unroll
    for (int t = 0; t < 8; ++t) {
        int n = t * 16 + m;
        float c0 = cw[n], t0 = tw[n], b1 = bc[n], b2 = bf[n];
#pragma unroll
        for (int i = 0; i < 8; ++i) {
            float r = 0.5f * ((cc[t][i] + b1) * c0 + (cf[t][i] + b2) * t0);
            out[(size_t)(rbase + i) * DFEAT + n] = r;
        }
    }
}

// ---------------- LayerNorm + ReLU (in-place on out), one block per row -------
__global__ __launch_bounds__(128) void k_ln(float* __restrict__ out,
                                            const float* __restrict__ gamma,
                                            const float* __restrict__ beta) {
    int row = blockIdx.x;
    int d = threadIdx.x;              // 128 threads = 4 waves
    float x = out[(size_t)row * DFEAT + d];
    float s = x, q = x * x;
#pragma unroll
    for (int off = 16; off >= 1; off >>= 1) {
        s += __shfl_xor(s, off, 32);
        q += __shfl_xor(q, off, 32);
    }
    __shared__ float ss[4], qq[4];
    int w = threadIdx.x >> 5;
    if ((threadIdx.x & 31) == 0) { ss[w] = s; qq[w] = q; }
    __syncthreads();
    float S = ss[0] + ss[1] + ss[2] + ss[3];
    float Q = qq[0] + qq[1] + qq[2] + qq[3];
    float mu = S * (1.f / DFEAT);
    float var = Q * (1.f / DFEAT) - mu * mu;
    float y = (x - mu) * rsqrtf(var + LN_EPS) * gamma[d] + beta[d];
    out[(size_t)row * DFEAT + d] = fmaxf(y, 0.f);
}

extern "C" void kernel_launch(void* const* d_in, const int* in_sizes, int n_in,
                              void* d_out, int out_size, void* d_ws, size_t ws_size,
                              hipStream_t stream) {
    const float* curr_h   = (const float*)d_in[0];
    const float* next_h   = (const float*)d_in[1];
    const float* curr_inc = (const float*)d_in[2];
    const float* Wc       = (const float*)d_in[3];
    const float* bc       = (const float*)d_in[4];
    const float* Wf       = (const float*)d_in[5];
    const float* bf       = (const float*)d_in[6];
    const float* conv_w   = (const float*)d_in[7];
    const float* td_w     = (const float*)d_in[8];
    const float* gamma    = (const float*)d_in[9];
    const float* beta     = (const float*)d_in[10];
    const int*   esrc     = (const int*)d_in[11];
    const int*   edst     = (const int*)d_in[12];
    float* out = (float*)d_out;

    // workspace carve-up (bytes)
    char* ws = (char*)d_ws;
    size_t o = 0;
    float* degO = (float*)(ws + o); o += (size_t)N_NODES * 4;
    float* degI = (float*)(ws + o); o += (size_t)N_NODES * 4;
    float* rsO  = (float*)(ws + o); o += (size_t)N_NODES * 4;
    float* rsI  = (float*)(ws + o); o += (size_t)N_NODES * 4;
    unsigned short* bT = (unsigned short*)(ws + o); o += (size_t)M_NODES * DFEAT * 2;
    float* fused = (float*)(ws + o); o += (size_t)N_NODES * DFEAT * 4;
    float* acc1  = (float*)(ws + o); o += (size_t)N_NODES * DFEAT * 4;
    float* acc2  = (float*)(ws + o); o += (size_t)N_NODES * DFEAT * 4;
    (void)ws_size; (void)o; (void)in_sizes; (void)n_in; (void)out_size;

    // 1) degrees
    k_zero<<<(2 * N_NODES + 255) / 256, 256, 0, stream>>>(degO, 2 * N_NODES);
    k_deg<<<(NEDGE + 255) / 256, 256, 0, stream>>>(esrc, edst, degO, degI);
    k_rs<<<(N_NODES + 255) / 256, 256, 0, stream>>>(degO, degI, rsO, rsI);
    // 2) next_h -> bf16 transposed
    k_bT<<<(M_NODES * DFEAT) / 256, 256, 0, stream>>>(next_h, bT);
    // 3) dominant bf16 WMMA GEMM: 128 blocks x 8 waves, async-LDS staged B
    k_dense<<<N_NODES / 128, 256, 0, stream>>>(curr_inc, bT, fused);
    // 4) SpMM (self-loop init + atomic edge scatter)
    k_initacc<<<(N_NODES * DFEAT) / 256, 256, 0, stream>>>(curr_h, fused, rsO, acc1, acc2);
    k_scatter<<<((size_t)NEDGE * DFEAT) / 256, 256, 0, stream>>>(esrc, edst, curr_h, fused,
                                                                 rsO, acc1, acc2);
    // 5) projection GEMMs (f32 WMMA) + combine
    k_out<<<(N_NODES / 16) * 32 / 256, 256, 0, stream>>>(acc1, acc2, rsI, Wc, bc, Wf, bf,
                                                         conv_w, td_w, out);
    // 6) LayerNorm + ReLU in place
    k_ln<<<N_NODES, 128, 0, stream>>>(out, gamma, beta);
}
